// PAB_CAB_91216515432883
// MI455X (gfx1250) — compile-verified, matrix operand order checked by reading, not run on hardware
//
#include <hip/hip_runtime.h>

// ---------------------------------------------------------------------------
// Channel-attention block (Restormer-style) for MI455X / gfx1250.
// B=2, C=384, H=W=128 (N=16384), 8 heads, d=48.
// 1x1 convs -> bf16 WMMA GEMMs (v_wmma_f32_16x16x32_bf16, f32 accumulate),
//              4 WMMA tiles per wave per K-step (16M x 64N strip).
// q.k^T      -> fp32 WMMA (v_wmma_f32_16x16x4_f32), dual accumulator chains.
// attn tile  -> LDS via global_load_async_to_lds_b128 (+ s_wait_asynccnt).
// ---------------------------------------------------------------------------

#define CDIM   384
#define BDIM   2
#define HDIM   128
#define WDIM   128
#define NDIM   (HDIM * WDIM)      // 16384
#define NHEADS 8
#define DHEAD  (CDIM / NHEADS)    // 48

typedef __attribute__((ext_vector_type(16))) __bf16 v16bf;
typedef __attribute__((ext_vector_type(8)))  float  v8f;
typedef __attribute__((ext_vector_type(2)))  float  v2f;

union FragBF { v16bf v; unsigned int u[8]; };

__device__ __forceinline__ unsigned short f2bf(float f) {
    union { float f; unsigned int u; } x; x.f = f;
    unsigned int r = x.u + 0x7FFFu + ((x.u >> 16) & 1u);   // round-to-nearest-even
    return (unsigned short)(r >> 16);
}
__device__ __forceinline__ unsigned int pack_bf2(float lo, float hi) {
    return (unsigned int)f2bf(lo) | ((unsigned int)f2bf(hi) << 16);
}

// ---------------------------------------------------------------------------
// 1x1-conv GEMM: Y[b, m, n] = sum_k W[m, k] * X[b, k, n]
// Block = 256 threads = 8 waves (2 mw x 4 nw). Block tile: 32 M x 256 N.
// K-step 32. Each wave: one A fragment, four B fragments, four
// v_wmma_f32_16x16x32_bf16 per K-step (A reuse amortizes LDS traffic).
// ---------------------------------------------------------------------------
__global__ __launch_bounds__(256) void conv1x1_wmma(
    const float* __restrict__ X, const float* __restrict__ W,
    float* __restrict__ Y, int M, int K, int N,
    long in_bstride, long out_bstride)
{
    __shared__ unsigned short As[32][36];    // [m][k]  bf16 bits (pad 36)
    __shared__ unsigned short Bs[256][36];   // [n][k]  bf16 bits, transposed

    const int tN = blockIdx.x * 256;
    const int tM = blockIdx.y * 32;
    const float* Xb = X + (long)blockIdx.z * in_bstride;
    float*       Yb = Y + (long)blockIdx.z * out_bstride;

    const int t    = threadIdx.x;
    const int wave = t >> 5;
    const int lane = t & 31;
    const int mw   = wave >> 2;              // 0..1
    const int nw   = wave & 3;               // 0..3

    v8f acc0 = {0,0,0,0,0,0,0,0};
    v8f acc1 = {0,0,0,0,0,0,0,0};
    v8f acc2 = {0,0,0,0,0,0,0,0};
    v8f acc3 = {0,0,0,0,0,0,0,0};

    const int row = mw * 16 + (lane & 15);
    const int g   = (lane < 16) ? 0 : 4;     // dword offset within a k-row

    for (int k0 = 0; k0 < K; k0 += 32) {
        {   // stage A tile (32 m x 32 k): 4 elems/thread, packed dword stores
            int r = t >> 3;
            int c = (t & 7) * 4;
            const float* src = W + (long)(tM + r) * K + k0 + c;
            unsigned int* dst = (unsigned int*)&As[r][0];
            dst[(c >> 1) + 0] = pack_bf2(src[0], src[1]);
            dst[(c >> 1) + 1] = pack_bf2(src[2], src[3]);
        }
        {   // stage B tile (32 k x 256 n) -> transposed LDS; thread = column
            const float* src = Xb + (long)k0 * N + tN + t;
            unsigned int* dst = (unsigned int*)&Bs[t][0];
            #pragma unroll
            for (int r = 0; r < 32; r += 2) {
                float f0 = src[(long)r * N];
                float f1 = src[(long)(r + 1) * N];
                dst[r >> 1] = pack_bf2(f0, f1);
            }
            if (k0 + 32 < K) {   // prefetch next K-step (global_prefetch_b8)
                __builtin_prefetch(src + (long)32 * N, 0, 3);
                __builtin_prefetch(src + (long)48 * N, 0, 3);
            }
        }
        __syncthreads();

        FragBF a;
        const unsigned int* ar = (const unsigned int*)&As[row][0];
        #pragma unroll
        for (int j = 0; j < 4; ++j) {
            a.u[j]     = ar[g + j];          // kk = (lane<16?0:8)  + 2j,2j+1
            a.u[4 + j] = ar[8 + g + j];      // kk = (lane<16?16:24)+ 2j,2j+1
        }
        #pragma unroll
        for (int s = 0; s < 4; ++s) {
            FragBF b;
            const int col = nw * 64 + s * 16 + (lane & 15);
            const unsigned int* br = (const unsigned int*)&Bs[col][0];
            #pragma unroll
            for (int j = 0; j < 4; ++j) {
                b.u[j]     = br[g + j];
                b.u[4 + j] = br[8 + g + j];
            }
            v8f& acc = (s == 0) ? acc0 : (s == 1) ? acc1 : (s == 2) ? acc2 : acc3;
            acc = __builtin_amdgcn_wmma_f32_16x16x32_bf16(
                      false, a.v, false, b.v, (short)0, acc, false, false);
        }
        __syncthreads();
    }

    // C/D layout: VGPR r -> M = r (lanes 0-15) / r+8 (lanes 16-31), N = lane&15
    const int rbase = tM + mw * 16 + ((lane < 16) ? 0 : 8);
    #pragma unroll
    for (int s = 0; s < 4; ++s) {
        const v8f& acc = (s == 0) ? acc0 : (s == 1) ? acc1 : (s == 2) ? acc2 : acc3;
        const int n = tN + nw * 64 + s * 16 + (lane & 15);
        #pragma unroll
        for (int r = 0; r < 8; ++r)
            Yb[(long)(rbase + r) * N + n] = acc[r];
    }
}

// ---------------------------------------------------------------------------
// Depthwise 3x3, SAME padding (cross-correlation, matching lax.conv).
// ---------------------------------------------------------------------------
__global__ __launch_bounds__(256) void dwconv3x3(
    const float* __restrict__ X, const float* __restrict__ W9,
    float* __restrict__ Y, int Cch, long in_bstride, long out_bstride, long total)
{
    long idx = (long)blockIdx.x * 256 + threadIdx.x;
    if (idx >= total) return;
    int x = (int)(idx % WDIM); long r = idx / WDIM;
    int y = (int)(r % HDIM);   r /= HDIM;
    int c = (int)(r % Cch);
    int b = (int)(r / Cch);

    const float* in = X + (long)b * in_bstride + (long)c * NDIM;
    const float* w  = W9 + c * 9;
    float acc = 0.f;
    #pragma unroll
    for (int dy = -1; dy <= 1; ++dy) {
        int yy = y + dy;
        if (yy < 0 || yy >= HDIM) continue;
        #pragma unroll
        for (int dx = -1; dx <= 1; ++dx) {
            int xx = x + dx;
            if (xx < 0 || xx >= WDIM) continue;
            acc += in[yy * WDIM + xx] * w[(dy + 1) * 3 + (dx + 1)];
        }
    }
    Y[(long)b * out_bstride + (long)c * NDIM + y * WDIM + x] = acc;
}

// qmix = sigmoid(g)*q1 + (1-sigmoid(g))*q2   (in-place on g is safe)
__global__ __launch_bounds__(256) void gate_blend(
    const float* __restrict__ g, const float* __restrict__ q1,
    const float* __restrict__ q2, float* __restrict__ out, long n)
{
    long i = (long)blockIdx.x * 256 + threadIdx.x;
    if (i >= n) return;
    float s = 1.f / (1.f + __expf(-g[i]));
    out[i] = s * q1[i] + (1.f - s) * q2[i];
}

// L2-normalize each row of length N; row (b, c) at base + b*bstride + c*N.
__global__ __launch_bounds__(256) void l2norm_rows(
    float* __restrict__ X, int rows_per_batch, long bstride, int N)
{
    int row = blockIdx.x;
    int b = row / rows_per_batch, c = row % rows_per_batch;
    float* p = X + (long)b * bstride + (long)c * N;
    __shared__ float red[256];
    float s = 0.f;
    for (int i = threadIdx.x; i < N; i += 256) { float v = p[i]; s += v * v; }
    red[threadIdx.x] = s;
    __syncthreads();
    for (int o = 128; o > 0; o >>= 1) {
        if (threadIdx.x < o) red[threadIdx.x] += red[threadIdx.x + o];
        __syncthreads();
    }
    float inv = 1.f / fmaxf(sqrtf(red[0]), 1e-12f);
    for (int i = threadIdx.x; i < N; i += 256) p[i] *= inv;
}

// ---------------------------------------------------------------------------
// attn[b,h,c,d] = sum_n q[b, h*48+c, n] * k[b, h*48+d, n]
// One wave per 16x16 tile; grid (3, 3, B*NHEADS). fp32 WMMA 16x16x4 over
// K=16384 with two independent accumulator chains.
// ---------------------------------------------------------------------------
__global__ __launch_bounds__(32) void attn_qk(
    const float* __restrict__ Q, const float* __restrict__ Kt,
    float* __restrict__ A, int N)
{
    const int bh = blockIdx.z, b = bh / NHEADS, h = bh % NHEADS;
    const float* q = Q + (long)b * CDIM * N      + (long)h * DHEAD * N;
    const float* k = Kt + (long)b * 2 * CDIM * N + (long)h * DHEAD * N;
    const int lane = threadIdx.x;
    float out8[8];

#if __has_builtin(__builtin_amdgcn_wmma_f32_16x16x4_f32)
    const int mrow = blockIdx.x * 16 + (lane & 15);
    const int ncol = blockIdx.y * 16 + (lane & 15);
    const int kofs = (lane < 16) ? 0 : 2;
    const float* qp = q + (long)mrow * N + kofs;
    const float* kp = k + (long)ncol * N + kofs;
    v8f acc0 = {0,0,0,0,0,0,0,0};
    v8f acc1 = {0,0,0,0,0,0,0,0};
    for (int n0 = 0; n0 < N; n0 += 8) {      // two independent WMMA chains
        v2f a0 = { qp[n0],     qp[n0 + 1] };
        v2f b0 = { kp[n0],     kp[n0 + 1] };
        v2f a1 = { qp[n0 + 4], qp[n0 + 5] };
        v2f b1 = { kp[n0 + 4], kp[n0 + 5] };
        acc0 = __builtin_amdgcn_wmma_f32_16x16x4_f32(
                   false, a0, false, b0, (short)0, acc0, false, false);
        acc1 = __builtin_amdgcn_wmma_f32_16x16x4_f32(
                   false, a1, false, b1, (short)0, acc1, false, false);
    }
    #pragma unroll
    for (int r = 0; r < 8; ++r) out8[r] = acc0[r] + acc1[r];
#else
    // scalar fallback producing identical lane/row mapping
    #pragma unroll
    for (int r = 0; r < 8; ++r) {
        int rowi = blockIdx.x * 16 + r + ((lane < 16) ? 0 : 8);
        int coli = blockIdx.y * 16 + (lane & 15);
        const float* qp = q + (long)rowi * N;
        const float* kp = k + (long)coli * N;
        float acc = 0.f;
        for (int n = 0; n < N; ++n) acc += qp[n] * kp[n];
        out8[r] = acc;
    }
#endif
    float* o = A + (long)bh * DHEAD * DHEAD;
    const int col   = blockIdx.y * 16 + (lane & 15);
    const int rbase = blockIdx.x * 16 + ((lane < 16) ? 0 : 8);
    #pragma unroll
    for (int r = 0; r < 8; ++r) o[(rbase + r) * DHEAD + col] = out8[r];
}

// softmax over last dim (48) with temperature scaling; one thread per row.
__global__ __launch_bounds__(256) void softmax48(
    float* __restrict__ A, const float* __restrict__ temp, int rows)
{
    int r = blockIdx.x * 256 + threadIdx.x;
    if (r >= rows) return;
    int h = (r / DHEAD) % NHEADS;
    float t = temp[h];
    float* p = A + (long)r * DHEAD;
    float v[DHEAD], mx = -3.4e38f;
    #pragma unroll
    for (int j = 0; j < DHEAD; ++j) { v[j] = p[j] * t; mx = fmaxf(mx, v[j]); }
    float s = 0.f;
    #pragma unroll
    for (int j = 0; j < DHEAD; ++j) { v[j] = __expf(v[j] - mx); s += v[j]; }
    float inv = 1.f / s;
    #pragma unroll
    for (int j = 0; j < DHEAD; ++j) p[j] = v[j] * inv;
}

// out[b, h*48+c, n] = sum_d attn[b,h,c,d] * v[b, C + h*48+d, n]
// The 48x48 attn tile is pulled into LDS with async global->LDS b128 copies.
__global__ __launch_bounds__(256) void attn_v(
    const float* __restrict__ A, const float* __restrict__ V,
    float* __restrict__ O, int N)
{
    const int bh = blockIdx.y, b = bh / NHEADS, h = bh % NHEADS;
    __shared__ float a[DHEAD * DHEAD];          // 2304 floats = 576 x b128
    const float* src = A + (long)bh * DHEAD * DHEAD;
#if defined(__gfx1250__)
    if (threadIdx.x < 192) {
        #pragma unroll
        for (int i = 0; i < 3; ++i) {
            int ch = threadIdx.x * 3 + i;       // 192*3 = 576 chunks of 16B
            unsigned int lds = (unsigned int)(unsigned long long)(&a[ch * 4]);
            const float* gp = src + ch * 4;
            asm volatile("global_load_async_to_lds_b128 %0, %1, off"
                         :: "v"(lds), "v"(gp) : "memory");
        }
    }
    asm volatile("s_wait_asynccnt 0x0" ::: "memory");
    __syncthreads();
#else
    for (int i = threadIdx.x; i < DHEAD * DHEAD; i += 256) a[i] = src[i];
    __syncthreads();
#endif
    const int n = blockIdx.x * 256 + threadIdx.x;
    const float* v = V + (long)b * 2 * CDIM * N + (long)(CDIM + h * DHEAD) * N + n;
    float vr[DHEAD];
    #pragma unroll
    for (int d = 0; d < DHEAD; ++d) vr[d] = v[(long)d * N];
    float* o = O + (long)b * CDIM * N + (long)h * DHEAD * N + n;
    for (int c = 0; c < DHEAD; ++c) {
        float acc = 0.f;
        #pragma unroll
        for (int d = 0; d < DHEAD; ++d) acc += a[c * DHEAD + d] * vr[d];
        o[(long)c * N] = acc;
    }
}

// ---------------------------------------------------------------------------
extern "C" void kernel_launch(void* const* d_in, const int* in_sizes, int n_in,
                              void* d_out, int out_size, void* d_ws, size_t ws_size,
                              hipStream_t stream)
{
    (void)in_sizes; (void)n_in; (void)out_size; (void)ws_size;
    const int C = CDIM, Bn = BDIM, N = NDIM;
    const long CN  = (long)C * N;          // per-batch plane (C channels)
    const long CNB = CN * Bn;              // full C-channel tensor

    const float* prompt    = (const float*)d_in[0];
    const float* x_mod     = (const float*)d_in[1];
    const float* y         = (const float*)d_in[2];
    const float* q1_w      = (const float*)d_in[3];
    const float* q2_w      = (const float*)d_in[4];
    const float* q_dw_w    = (const float*)d_in[5];
    const float* gate_dw_w = (const float*)d_in[6];
    const float* gate_pw_w = (const float*)d_in[7];
    const float* kv_w      = (const float*)d_in[8];
    const float* kv_dw_w   = (const float*)d_in[9];
    const float* proj_w    = (const float*)d_in[10];
    const float* temp      = (const float*)d_in[11];

    // workspace layout (floats): 6*CNB + attn  (~302 MB)
    float* ws    = (float*)d_ws;
    float* q1buf = ws;                 // [B,C,N]
    float* q2buf = ws + CNB;           // [B,C,N]   (later: attention output)
    float* t2a   = ws + 2 * CNB;       // [B,2C,N]  (gate-dw cat / final kv)
    float* t2b   = ws + 4 * CNB;       // [B,2C,N]  (kv-pw; first CNB doubles as gbuf)
    float* gbuf  = t2b;                // [B,C,N]
    float* attn  = ws + 6 * CNB;       // [B,H,48,48]

    const dim3 blk(256);
    const dim3 gemmC(N / 256, C / 32, Bn);         // M = C
    const dim3 gemm2C(N / 256, (2 * C) / 32, Bn);  // M = 2C
    const long totC  = CNB;
    const long tot2C = 2 * CNB;
    const int  gC  = (int)((totC  + 255) / 256);
    const int  g2C = (int)((tot2C + 255) / 256);

    // q1 = 1x1(prompt), q2 = 1x1(x_mod)
    conv1x1_wmma<<<gemmC, blk, 0, stream>>>(prompt, q1_w, q1buf, C, C, N, CN, CN);
    conv1x1_wmma<<<gemmC, blk, 0, stream>>>(x_mod,  q2_w, q2buf, C, C, N, CN, CN);

    // gate_dw on cat(q1,q2): two depthwise passes into the 2C tensor
    dwconv3x3<<<gC, blk, 0, stream>>>(q1buf, gate_dw_w,         t2a,               C, CN, 2 * CN, totC);
    dwconv3x3<<<gC, blk, 0, stream>>>(q2buf, gate_dw_w + C * 9, t2a + C * (long)N, C, CN, 2 * CN, totC);

    // gate = 1x1(gate_dw_out), blend, then q depthwise
    conv1x1_wmma<<<gemmC, blk, 0, stream>>>(t2a, gate_pw_w, gbuf, C, 2 * C, N, 2 * CN, CN);
    gate_blend<<<gC, blk, 0, stream>>>(gbuf, q1buf, q2buf, gbuf, totC);
    dwconv3x3<<<gC, blk, 0, stream>>>(gbuf, q_dw_w, q1buf, C, CN, CN, totC);

    // kv = dw3x3(1x1(y))
    conv1x1_wmma<<<gemm2C, blk, 0, stream>>>(y, kv_w, t2b, 2 * C, C, N, CN, 2 * CN);
    dwconv3x3<<<g2C, blk, 0, stream>>>(t2b, kv_dw_w, t2a, 2 * C, 2 * CN, 2 * CN, tot2C);

    // l2norm(q), l2norm(k)  (k = first C channels of t2a)
    l2norm_rows<<<Bn * C, blk, 0, stream>>>(q1buf, C, CN, N);
    l2norm_rows<<<Bn * C, blk, 0, stream>>>(t2a,   C, 2 * CN, N);

    // channel attention
    attn_qk<<<dim3(3, 3, Bn * NHEADS), dim3(32), 0, stream>>>(q1buf, t2a, attn, N);
    softmax48<<<(Bn * NHEADS * DHEAD + 255) / 256, blk, 0, stream>>>(attn, temp, Bn * NHEADS * DHEAD);
    attn_v<<<dim3(N / 256, Bn * NHEADS), blk, 0, stream>>>(attn, t2a, q2buf, N);

    // final projection straight into d_out
    conv1x1_wmma<<<gemmC, blk, 0, stream>>>(q2buf, proj_w, (float*)d_out, C, C, N, CN, CN);
}
